// LlamaBlock_71219147702548
// MI455X (gfx1250) — matrix operand order repaired and checked by reading, not compiled, not run on hardware
//
#include <hip/hip_runtime.h>

// ---------- types ----------
typedef __attribute__((ext_vector_type(16))) __bf16      v16bf;
typedef __attribute__((ext_vector_type(8)))  float       v8f;
typedef __attribute__((ext_vector_type(4)))  unsigned int u32x4;
typedef unsigned short ushort_t;

union Frag {                 // one WMMA 16-bit A/B operand (16 bf16 per lane)
    v16bf    bf;
    u32x4    q[2];
    ushort_t u[16];
};

__device__ __forceinline__ v8f vzero8() {
    v8f z = {0.f, 0.f, 0.f, 0.f, 0.f, 0.f, 0.f, 0.f};
    return z;
}

__device__ __forceinline__ ushort_t f2bf(float f) {
    unsigned int u = __float_as_uint(f);
    u += 0x7FFFu + ((u >> 16) & 1u);        // round-to-nearest-even
    return (ushort_t)(u >> 16);
}

__device__ __forceinline__ v8f wmma_bf16(const Frag& a, const Frag& b, v8f c) {
    // D(16x16,f32) = A(16x32,bf16) * B(32x16,bf16) + C
    return __builtin_amdgcn_wmma_f32_16x16x32_bf16(
        /*neg_a=*/false, a.bf, /*neg_b=*/false, b.bf,
        /*c_mod=*/(short)0, c, /*reuse_a=*/false, /*reuse_b=*/false);
}

// ---------- problem constants ----------
constexpr int Bc   = 2;
constexpr int Sc   = 2048;
constexpr int DIMc = 2048;
constexpr int Hc   = 16;
constexpr int HDc  = 128;
constexpr int Mg   = Bc * Sc;     // 4096 GEMM rows
constexpr int Ng   = Hc * HDc;    // 2048 GEMM cols
constexpr int Kg   = DIMc;        // 2048 GEMM depth

// ============================================================
// Kernel 1a: f32 -> bf16 (hidden states, row-major unchanged)
// ============================================================
__global__ void cvt_hs_kernel(const float* __restrict__ in,
                              ushort_t* __restrict__ out, int n) {
    for (int i = blockIdx.x * blockDim.x + threadIdx.x; i < n;
         i += gridDim.x * blockDim.x)
        out[i] = f2bf(in[i]);
}

// ============================================================
// Kernel 1b: weight f32 [K][N] -> bf16 transposed Wt[N][K]
// ============================================================
__global__ void cvt_wT_kernel(const float* __restrict__ W,
                              ushort_t* __restrict__ Wt) {
    for (int i = blockIdx.x * blockDim.x + threadIdx.x; i < Kg * Ng;
         i += gridDim.x * blockDim.x) {
        int k = i >> 11;            // / Ng
        int n = i & (Ng - 1);
        Wt[(size_t)n * Kg + k] = f2bf(W[i]);
    }
}

// ============================================================
// Kernel 2: QKV projection GEMM, bf16 WMMA, f32 accumulate
//   X[M=4096][K=2048] x Wt[N=2048][K]  -> bf16 outputs
//   z=0: Q[b,h,s,hd]  z=1: K[b,h,s,hd]  z=2: V^T[b,h,hd,s]
// Block: 128 thr (4 waves, 2x2), tile 128x128, wave tile 64x64,
// K-step 32 through double-buffered LDS (one barrier per step).
// ============================================================
__global__ __launch_bounds__(128)
void qkv_gemm_kernel(const ushort_t* __restrict__ X,
                     const ushort_t* __restrict__ Wt0,
                     const ushort_t* __restrict__ Wt1,
                     const ushort_t* __restrict__ Wt2,
                     ushort_t* __restrict__ Qo,
                     ushort_t* __restrict__ Ko,
                     ushort_t* __restrict__ Vto) {
    const int m0 = blockIdx.x * 128;
    const int n0 = blockIdx.y * 128;
    const int z  = blockIdx.z;
    const ushort_t* Wt = (z == 0) ? Wt0 : ((z == 1) ? Wt1 : Wt2);

    const int tid  = threadIdx.x;
    const int lane = tid & 31;
    const int w    = tid >> 5;
    const int wm   = w >> 1;          // 0..1 -> 64-row slab
    const int wn   = w & 1;           // 0..1 -> 64-col slab
    const int lm   = lane & 15;
    const int akb  = (lane < 16) ? 0 : 8;    // A frag K base (ISA 16-bit A layout)
    const int bkb  = (lane < 16) ? 0 : 8;    // B frag K base

    __shared__ __align__(16) ushort_t Xs[2][128 * 32];   // 2 x 8 KB
    __shared__ __align__(16) ushort_t Ws[2][128 * 32];   // 2 x 8 KB

    v8f acc[4][4];
#pragma unroll
    for (int mi = 0; mi < 4; ++mi)
#pragma unroll
        for (int ni = 0; ni < 4; ++ni) acc[mi][ni] = vzero8();

    // ---- prologue: stage K-tile 0 into buffer 0 ----
    u32x4 rx[4], rw[4];
#pragma unroll
    for (int i = 0; i < 4; ++i) {
        int c = tid + i * 128, r = c >> 2, cc = c & 3;
        rx[i] = *(const u32x4*)(X  + ((size_t)(m0 + r) * Kg + cc * 8));
        rw[i] = *(const u32x4*)(Wt + ((size_t)(n0 + r) * Kg + cc * 8));
    }
#pragma unroll
    for (int i = 0; i < 4; ++i) {
        int c = tid + i * 128, r = c >> 2, cc = c & 3;
        *(u32x4*)&Xs[0][r * 32 + cc * 8] = rx[i];
        *(u32x4*)&Ws[0][r * 32 + cc * 8] = rw[i];
    }
    __syncthreads();

    const int nIter = Kg / 32;        // 64
    for (int it = 0; it < nIter; ++it) {
        const int  cur     = it & 1;
        const bool hasNext = (it + 1 < nIter);

        // issue next tile's global loads first (latency hidden by WMMA chain)
        if (hasNext) {
            const int k0n = (it + 1) * 32;
#pragma unroll
            for (int i = 0; i < 4; ++i) {
                int c = tid + i * 128, r = c >> 2, cc = c & 3;
                rx[i] = *(const u32x4*)(X  + ((size_t)(m0 + r) * Kg + k0n + cc * 8));
                rw[i] = *(const u32x4*)(Wt + ((size_t)(n0 + r) * Kg + k0n + cc * 8));
            }
        }

        // fragments from current buffer
        Frag a[4], b[4];
#pragma unroll
        for (int mi = 0; mi < 4; ++mi) {
            const ushort_t* p = &Xs[cur][(wm * 64 + mi * 16 + lm) * 32 + akb];
            a[mi].q[0] = *(const u32x4*)p;          // K akb..akb+7
            a[mi].q[1] = *(const u32x4*)(p + 16);   // K akb+16..akb+23
        }
#pragma unroll
        for (int ni = 0; ni < 4; ++ni) {
            const ushort_t* p = &Ws[cur][(wn * 64 + ni * 16 + lm) * 32 + bkb];
            b[ni].q[0] = *(const u32x4*)p;          // K bkb..bkb+7
            b[ni].q[1] = *(const u32x4*)(p + 16);   // K bkb+16..bkb+23
        }
        // 16 WMMAs: each A fragment reused across 4 B fragments
#pragma unroll
        for (int mi = 0; mi < 4; ++mi)
#pragma unroll
            for (int ni = 0; ni < 4; ++ni)
                acc[mi][ni] = wmma_bf16(a[mi], b[ni], acc[mi][ni]);

        // drain next tile into the other buffer
        if (hasNext) {
#pragma unroll
            for (int i = 0; i < 4; ++i) {
                int c = tid + i * 128, r = c >> 2, cc = c & 3;
                *(u32x4*)&Xs[cur ^ 1][r * 32 + cc * 8] = rx[i];
                *(u32x4*)&Ws[cur ^ 1][r * 32 + cc * 8] = rw[i];
            }
        }
        __syncthreads();
    }

    // epilogue: C layout — VGPR g: rows g (lanes 0-15) / g+8 (lanes 16-31), col = lm
#pragma unroll
    for (int mi = 0; mi < 4; ++mi)
#pragma unroll
        for (int ni = 0; ni < 4; ++ni)
#pragma unroll
            for (int g = 0; g < 8; ++g) {
                int m = m0 + wm * 64 + mi * 16 + ((lane < 16) ? g : (g + 8));
                int n = n0 + wn * 64 + ni * 16 + lm;
                ushort_t v = f2bf(acc[mi][ni][g]);
                int bb = m >> 11, s = m & (Sc - 1);
                int h  = n >> 7,  hd = n & (HDc - 1);
                size_t bh = (size_t)bb * Hc + h;
                if (z == 0)      Qo [(bh * Sc + s) * HDc + hd] = v;
                else if (z == 1) Ko [(bh * Sc + s) * HDc + hd] = v;
                else             Vto[(bh * HDc + hd) * Sc + s] = v;
            }
}

// ============================================================
// Kernel 3: causal flash attention, bf16 WMMA, f32 softmax/acc
// Block = (q-tile of 128, head, batch); 8 waves x 16 q-rows.
// KV chunks of 32 double-buffered in LDS (one barrier per step).
// ============================================================
__global__ __launch_bounds__(256)
void attn_kernel(const ushort_t* __restrict__ Qb,
                 const ushort_t* __restrict__ Kb,
                 const ushort_t* __restrict__ Vt,
                 float* __restrict__ out) {
    const int q0   = blockIdx.x * 128;
    const int h    = blockIdx.y;
    const int b    = blockIdx.z;
    const int tid  = threadIdx.x;
    const int lane = tid & 31;
    const int w    = tid >> 5;
    const int lm   = lane & 15;
    const int akb  = (lane < 16) ? 0 : 8;
    const int bkb  = (lane < 16) ? 0 : 8;

    const size_t bh = (size_t)b * Hc + h;
    const ushort_t* Qp = Qb + bh * Sc * HDc;   // [S][HD]
    const ushort_t* Kp = Kb + bh * Sc * HDc;   // [S][HD]
    const ushort_t* Vp = Vt + bh * HDc * Sc;   // [HD][S]

    __shared__ __align__(16) ushort_t Ks[2][32 * 128];   // 2 x 8 KB  [kv][hd]
    __shared__ __align__(16) ushort_t Vs[2][128 * 32];   // 2 x 8 KB  [hd][kv]
    __shared__ __align__(16) ushort_t Ps[8 * 16 * 32];   // 8 KB per-wave P tiles

    // ---- load this wave's Q fragments (16 rows x 128 hd) once ----
    Frag qf[4];
    const int qr = q0 + (w << 4) + lm;
#pragma unroll
    for (int kc = 0; kc < 4; ++kc) {
        const ushort_t* p = Qp + (size_t)qr * HDc + kc * 32 + akb;
        qf[kc].q[0] = *(const u32x4*)p;
        qf[kc].q[1] = *(const u32x4*)(p + 16);
    }

    v8f o[8];
#pragma unroll
    for (int nt = 0; nt < 8; ++nt) o[nt] = vzero8();
    float mrow[8], lrow[8];
#pragma unroll
    for (int g = 0; g < 8; ++g) { mrow[g] = -1e30f; lrow[g] = 0.f; }

    const float scale = 0.08838834764831845f;   // 1/sqrt(128)
    ushort_t* pp = &Ps[w * 512];                // wave-private 16x32 P tile

    // ---- prologue: stage KV chunk 0 into buffer 0 ----
    u32x4 rk[2], rv[2];
#pragma unroll
    for (int i = 0; i < 2; ++i) {
        int c = tid + i * 256;
        rk[i] = *(const u32x4*)(Kp + (size_t)(c >> 4) * HDc + (c & 15) * 8);
        rv[i] = *(const u32x4*)(Vp + (size_t)(c >> 2) * Sc + (c & 3) * 8);
    }
#pragma unroll
    for (int i = 0; i < 2; ++i) {
        int c = tid + i * 256;
        *(u32x4*)&Ks[0][(c >> 4) * 128 + (c & 15) * 8] = rk[i];
        *(u32x4*)&Vs[0][(c >> 2) * 32  + (c & 3)  * 8] = rv[i];
    }
    __syncthreads();

    const int nIter = (q0 + 128) >> 5;          // causal bound for this q-tile
    for (int it = 0; it < nIter; ++it) {
        const int  j       = it << 5;
        const int  cur     = it & 1;
        const bool hasNext = (it + 1 < nIter);

        // issue next KV chunk's global loads (hidden under 16 WMMAs + softmax)
        if (hasNext) {
            const int jn = j + 32;
#pragma unroll
            for (int i = 0; i < 2; ++i) {
                int c = tid + i * 256;
                rk[i] = *(const u32x4*)(Kp + (size_t)(jn + (c >> 4)) * HDc + (c & 15) * 8);
                rv[i] = *(const u32x4*)(Vp + (size_t)(c >> 2) * Sc + jn + (c & 3) * 8);
            }
        }

        // ---- scores: S(16x32) = Q(16x128) . K^T(128x32), two 16x16 tiles ----
        v8f s0 = vzero8(), s1 = vzero8();
#pragma unroll
        for (int kc = 0; kc < 4; ++kc) {
            Frag k0f, k1f;
            const ushort_t* p0 = &Ks[cur][(0 * 16 + lm) * 128 + kc * 32 + bkb];
            const ushort_t* p1 = &Ks[cur][(1 * 16 + lm) * 128 + kc * 32 + bkb];
            k0f.q[0] = *(const u32x4*)p0; k0f.q[1] = *(const u32x4*)(p0 + 16);
            k1f.q[0] = *(const u32x4*)p1; k1f.q[1] = *(const u32x4*)(p1 + 16);
            s0 = wmma_bf16(qf[kc], k0f, s0);
            s1 = wmma_bf16(qf[kc], k1f, s1);
        }

        // ---- online softmax (per-row, rows live across 16-lane halves) ----
        float p0e[8], p1e[8];
#pragma unroll
        for (int g = 0; g < 8; ++g) {
            int r  = q0 + (w << 4) + ((lane < 16) ? g : (g + 8));
            int c0 = j + lm;
            float x0 = s0[g] * scale + ((c0      > r) ? -1e9f : 0.f);
            float x1 = s1[g] * scale + ((c0 + 16 > r) ? -1e9f : 0.f);
            float rm = fmaxf(x0, x1);
            rm = fmaxf(rm, __shfl_xor(rm, 1));
            rm = fmaxf(rm, __shfl_xor(rm, 2));
            rm = fmaxf(rm, __shfl_xor(rm, 4));
            rm = fmaxf(rm, __shfl_xor(rm, 8));
            float mn = fmaxf(mrow[g], rm);
            float e0 = __expf(x0 - mn);
            float e1 = __expf(x1 - mn);
            float rs = e0 + e1;
            rs += __shfl_xor(rs, 1);
            rs += __shfl_xor(rs, 2);
            rs += __shfl_xor(rs, 4);
            rs += __shfl_xor(rs, 8);
            float corr = __expf(mrow[g] - mn);
            lrow[g] = lrow[g] * corr + rs;
            mrow[g] = mn;
#pragma unroll
            for (int nt = 0; nt < 8; ++nt) o[nt][g] *= corr;
            p0e[g] = e0; p1e[g] = e1;
        }

        // ---- re-layout P: C-layout -> A-layout via wave-private LDS ----
#pragma unroll
        for (int g = 0; g < 8; ++g) {
            int r = (lane < 16) ? g : (g + 8);
            pp[r * 32 + lm]      = f2bf(p0e[g]);
            pp[r * 32 + 16 + lm] = f2bf(p1e[g]);
        }
        asm volatile("s_wait_dscnt 0" ::: "memory");   // same-wave DS RAW
        Frag af;
        {
            const ushort_t* p = &pp[lm * 32 + akb];
            af.q[0] = *(const u32x4*)p;
            af.q[1] = *(const u32x4*)(p + 16);
        }

        // ---- O += P(16x32) . V(32x128) ----
#pragma unroll
        for (int nt = 0; nt < 8; ++nt) {
            Frag vf;
            const ushort_t* p = &Vs[cur][(nt * 16 + lm) * 32 + bkb];
            vf.q[0] = *(const u32x4*)p;
            vf.q[1] = *(const u32x4*)(p + 16);
            o[nt] = wmma_bf16(af, vf, o[nt]);
        }

        // drain next KV chunk into the other buffer
        if (hasNext) {
#pragma unroll
            for (int i = 0; i < 2; ++i) {
                int c = tid + i * 256;
                *(u32x4*)&Ks[cur ^ 1][(c >> 4) * 128 + (c & 15) * 8] = rk[i];
                *(u32x4*)&Vs[cur ^ 1][(c >> 2) * 32  + (c & 3)  * 8] = rv[i];
            }
        }
        __syncthreads();
    }

    // ---- epilogue: O / l -> out[b,h,s,hd] (f32) ----
#pragma unroll
    for (int g = 0; g < 8; ++g) {
        int r = q0 + (w << 4) + ((lane < 16) ? g : (g + 8));
        float inv = 1.0f / lrow[g];
#pragma unroll
        for (int nt = 0; nt < 8; ++nt) {
            int cc = nt * 16 + lm;
            out[(bh * Sc + r) * (size_t)HDc + cc] = o[nt][g] * inv;
        }
    }
}

// ============================================================
// launcher
// ============================================================
extern "C" void kernel_launch(void* const* d_in, const int* in_sizes, int n_in,
                              void* d_out, int out_size, void* d_ws, size_t ws_size,
                              hipStream_t stream) {
    (void)in_sizes; (void)n_in; (void)out_size; (void)ws_size;
    const float* hidden = (const float*)d_in[0];
    // d_in[1] = attention_mask (causal additive); implemented analytically
    const float* Wq = (const float*)d_in[2];
    const float* Wk = (const float*)d_in[3];
    const float* Wv = (const float*)d_in[4];
    float* out = (float*)d_out;

    // workspace carve-up (bytes): total ~92.3 MB
    char* ws = (char*)d_ws;
    size_t off = 0;
    ushort_t* hsb = (ushort_t*)(ws + off); off += (size_t)Mg * Kg * 2;       // 16 MB
    ushort_t* wqt = (ushort_t*)(ws + off); off += (size_t)Ng * Kg * 2;       //  8 MB
    ushort_t* wkt = (ushort_t*)(ws + off); off += (size_t)Ng * Kg * 2;
    ushort_t* wvt = (ushort_t*)(ws + off); off += (size_t)Ng * Kg * 2;
    ushort_t* Qb  = (ushort_t*)(ws + off); off += (size_t)Bc * Hc * Sc * HDc * 2;
    ushort_t* Kb  = (ushort_t*)(ws + off); off += (size_t)Bc * Hc * Sc * HDc * 2;
    ushort_t* Vt  = (ushort_t*)(ws + off); off += (size_t)Bc * Hc * Sc * HDc * 2;

    // 1) precision conversion (bandwidth-trivial at 23.3 TB/s)
    cvt_hs_kernel<<<8192, 256, 0, stream>>>(hidden, hsb, Mg * Kg);
    cvt_wT_kernel<<<4096, 256, 0, stream>>>(Wq, wqt);
    cvt_wT_kernel<<<4096, 256, 0, stream>>>(Wk, wkt);
    cvt_wT_kernel<<<4096, 256, 0, stream>>>(Wv, wvt);

    // 2) QKV projection: grid (M/128, N/128, 3), 128 threads (4 waves)
    qkv_gemm_kernel<<<dim3(Mg / 128, Ng / 128, 3), 128, 0, stream>>>(
        hsb, wqt, wkt, wvt, Qb, Kb, Vt);

    // 3) causal flash attention: grid (S/128, H, B), 256 threads (8 waves)
    attn_kernel<<<dim3(Sc / 128, Hc, Bc), 256, 0, stream>>>(Qb, Kb, Vt, out);
}
